// SRGNN_44916767981584
// MI455X (gfx1250) — matrix operand order repaired and checked by reading, not compiled, not run on hardware
//
#include <hip/hip_runtime.h>
#include <hip/hip_bf16.h>
#include <math.h>

// Problem sizes (fixed by the reference)
constexpr int kB = 1024;    // batch
constexpr int kN = 50;      // nodes
constexpr int kS = 50;      // seq len
constexpr int kH = 128;     // hidden
constexpr int kV = 100000;  // vocab

typedef __attribute__((ext_vector_type(16))) __bf16 v16bf;
typedef __attribute__((ext_vector_type(8)))  __bf16 v8bf;
typedef __attribute__((ext_vector_type(8)))  float  v8f;

__device__ __forceinline__ __bf16 f2bf(float x) { return (__bf16)x; }
__device__ __forceinline__ float sigmoidf_(float x) { return 1.f / (1.f + __expf(-x)); }

// ---- WMMA fragment loaders (ISA 7.12.2, wave32) -------------------------
// A operand (16x32 bf16, row-major source, ld):
//   lanes 0-15: row m=lane,   K = kbase..+7 and kbase+16..+23
//   lanes 16-31: row m=lane-16, K shifted by +8
__device__ __forceinline__ v16bf load_a_frag(const __bf16* A, int ldA, int row0,
                                             int ks, int lane) {
  int m = lane & 15, hi = (lane >> 4) & 1;
  const __bf16* p = A + (size_t)(row0 + m) * ldA + ks * 32 + hi * 8;
  v16bf f;
  *((v8bf*)&f)       = *((const v8bf*)p);
  *(((v8bf*)&f) + 1) = *((const v8bf*)(p + 16));
  return f;
}

// B operand (32x16 bf16) for X@W^T, W row-major (Ncols x K):
//   lane n=lane&15 holds W row ntile*16+n; lanes 0-15: K kbase..+7 and kbase+16..+23,
//   lanes 16-31: K shifted by +8 (two 16B loads, same order as A).
__device__ __forceinline__ v16bf load_b_frag(const __bf16* W, int ldW, int ntile,
                                             int ks, int lane, int maxRow) {
  int n = lane & 15, hi = (lane >> 4) & 1;
  int wrow = ntile * 16 + n;
  wrow = wrow > maxRow ? maxRow : wrow;
  const __bf16* p = W + (size_t)wrow * ldW + ks * 32 + hi * 8;
  v16bf f;
  *((v8bf*)&f)       = *((const v8bf*)p);
  *(((v8bf*)&f) + 1) = *((const v8bf*)(p + 16));
  return f;
}

// ---- f32 -> bf16 conversion ---------------------------------------------
__global__ void cvt_bf16(const float* __restrict__ s, __bf16* __restrict__ d, size_t n) {
  size_t i = (size_t)blockIdx.x * blockDim.x + threadIdx.x;
  size_t stride = (size_t)gridDim.x * blockDim.x;
  for (; i < n; i += stride) d[i] = f2bf(s[i]);
}

// ---- zero fill (pad regions must be deterministic every call) -----------
__global__ void zero_u32(unsigned* __restrict__ d, size_t n) {
  size_t i = (size_t)blockIdx.x * blockDim.x + threadIdx.x;
  size_t stride = (size_t)gridDim.x * blockDim.x;
  for (; i < n; i += stride) d[i] = 0u;
}

// ---- h = emb[items], f32 + bf16 copies ----------------------------------
__global__ void gather_h(const float* __restrict__ emb, const int* __restrict__ items,
                         float* __restrict__ hF, __bf16* __restrict__ hB, int rows) {
  int i = blockIdx.x * blockDim.x + threadIdx.x;   // one per (row, 4 cols)
  int total = rows * (kH / 4);
  if (i >= total) return;
  int row = i >> 5;            // kH/4 == 32
  int c4  = (i & 31) * 4;
  int item = items[row];
  const float4 v = *((const float4*)(emb + (size_t)item * kH + c4));
  *((float4*)(hF + (size_t)row * kH + c4)) = v;
  __bf16* pb = hB + (size_t)row * kH + c4;
  pb[0] = f2bf(v.x); pb[1] = f2bf(v.y); pb[2] = f2bf(v.z); pb[3] = f2bf(v.w);
}

// ---- A -> zero-padded bf16 (B, 2 sides, 64 rows, 64 k) ------------------
__global__ void pad_A(const float* __restrict__ A, __bf16* __restrict__ Apad) {
  int i = blockIdx.x * blockDim.x + threadIdx.x;
  int total = kB * 2 * 64 * 64;
  if (i >= total) return;
  int k = i & 63, m = (i >> 6) & 63, side = (i >> 12) & 1, b = i >> 13;
  float v = (m < kN && k < kN)
          ? A[(size_t)b * (kN * 2 * kN) + m * (2 * kN) + side * kN + k] : 0.f;
  Apad[i] = f2bf(v);
}

// ---- Generic  out = X @ W^T (+bias)  with bf16 WMMA ----------------------
// X: (M,K) bf16 row-major, W: (N,K) bf16 row-major. K = KSTEPS*32.
// Block: MT row tiles (A-frags register-resident, shared by all waves);
// wave sweeps NT N-tiles, each B-frag load feeds MT WMMAs (B reuse = MT).
// NTST: non-temporal f32 stores (keep 410MB score stream out of L2).
// TPOSE: write outB transposed+batched for the graph matmul B operand:
//        row = b*kN + j  ->  outB[b*tStride + col*64 + j]   (j < kN only)
template <int KSTEPS, int MT, int NT, bool NTST, bool TPOSE>
__global__ void wmma_gemm_xwt(const __bf16* __restrict__ A, const __bf16* __restrict__ W,
                              const float* __restrict__ bias,
                              float* __restrict__ outF, __bf16* __restrict__ outB,
                              int M, int N, int tStride) {
  const int K = KSTEPS * 32;
  int lane = threadIdx.x & 31;
  int wave = threadIdx.x >> 5;
  int row0 = blockIdx.x * (MT * 16);
  if (row0 >= M) return;

  v16bf afrag[MT][KSTEPS];
#pragma unroll
  for (int r = 0; r < MT; ++r)
#pragma unroll
    for (int ks = 0; ks < KSTEPS; ++ks)
      afrag[r][ks] = load_a_frag(A, K, row0 + r * 16, ks, lane);

  int wavesPerBlock = blockDim.x >> 5;
  int ntBase = (blockIdx.y * wavesPerBlock + wave) * NT;
  int nl = lane & 15, hi = (lane >> 4) & 1;

  for (int t = 0; t < NT; ++t) {
    int ntile = ntBase + t;
    if (ntile * 16 >= N) return;

    // Prefetch next tile's W rows into cache (global_prefetch_b8 path).
    if (t + 1 < NT && (ntile + 1) * 16 < N) {
      int pr = (ntile + 1) * 16 + nl;
      pr = pr > N - 1 ? N - 1 : pr;
      __builtin_prefetch(W + (size_t)pr * K + hi * (K / 2), 0, 3);
    }

    v8f acc[MT];
#pragma unroll
    for (int r = 0; r < MT; ++r) acc[r] = v8f{};
#pragma unroll
    for (int ks = 0; ks < KSTEPS; ++ks) {
      v16bf b = load_b_frag(W, K, ntile, ks, lane, N - 1);
#pragma unroll
      for (int r = 0; r < MT; ++r)
        acc[r] = __builtin_amdgcn_wmma_f32_16x16x32_bf16(false, afrag[r][ks], false, b,
                                                         (short)0, acc[r], false, false);
    }

    int col = ntile * 16 + nl;
    if (col < N) {
      float bval = bias ? bias[col] : 0.f;
#pragma unroll
      for (int r = 0; r < MT; ++r) {
#pragma unroll
        for (int v = 0; v < 8; ++v) {
          int row = row0 + r * 16 + hi * 8 + v;
          float x = acc[r][v] + bval;
          if (outF) {
            if (NTST) __builtin_nontemporal_store(x, &outF[(size_t)row * N + col]);
            else      outF[(size_t)row * N + col] = x;
          }
          if (outB) {
            if (TPOSE) {
              int b_ = row / kN, j = row - b_ * kN;   // j < kN always (M = B*kN)
              outB[(size_t)b_ * tStride + col * 64 + j] = f2bf(x);
            } else {
              outB[(size_t)row * N + col] = f2bf(x);
            }
          }
        }
      }
    }
  }
}

// ---- Batched graph matmul: a_side = A_pad[:,side] @ hvT[:,side]^T --------
// All operands pre-padded -> pure vector loads + WMMA, no predication.
// Per block: one batch; 8 waves = 8 N-tiles over H; K = 64 (2 WMMA steps).
__global__ void wmma_graph_mm(const __bf16* __restrict__ Apad,  // (B,2,64,64)
                              const __bf16* __restrict__ hvT,   // (B,2,128,64)
                              const float* __restrict__ b_iah,
                              const float* __restrict__ b_oah,
                              __bf16* __restrict__ aio) {       // (B*50,256)
  int b = blockIdx.x;
  int lane = threadIdx.x & 31;
  int wave = threadIdx.x >> 5;  // N-tile over H: 0..7
  int nl = lane & 15, hi = (lane >> 4) & 1;
  int col = wave * 16 + nl;

#pragma unroll
  for (int side = 0; side < 2; ++side) {
    const __bf16* Ab = Apad + ((size_t)b * 2 + side) * 64 * 64;
    const __bf16* Bb = hvT  + ((size_t)b * 2 + side) * 128 * 64;
    const float* bias = side ? b_oah : b_iah;
    float bval = bias[col];

    v16bf bfrag[2];
#pragma unroll
    for (int ks = 0; ks < 2; ++ks) bfrag[ks] = load_b_frag(Bb, 64, wave, ks, lane, 127);

#pragma unroll
    for (int mt = 0; mt < 4; ++mt) {
      v16bf a0 = load_a_frag(Ab, 64, mt * 16, 0, lane);
      v16bf a1 = load_a_frag(Ab, 64, mt * 16, 1, lane);
      v8f c = {};
      c = __builtin_amdgcn_wmma_f32_16x16x32_bf16(false, a0, false, bfrag[0],
                                                  (short)0, c, false, false);
      c = __builtin_amdgcn_wmma_f32_16x16x32_bf16(false, a1, false, bfrag[1],
                                                  (short)0, c, false, false);
#pragma unroll
      for (int v = 0; v < 8; ++v) {
        int mo = mt * 16 + hi * 8 + v;
        if (mo < kN)
          aio[((size_t)b * kN + mo) * (2 * kH) + side * kH + col] = f2bf(c[v] + bval);
      }
    }
  }
}

// ---- Fused GRU cell: gi = aio@w_ih^T, gh = h@w_hh^T, gates, in-place h ---
__global__ void wmma_gru(const __bf16* __restrict__ aio,  // (R,256)
                         const __bf16* __restrict__ wih,  // (384,256)
                         const __bf16* __restrict__ whh,  // (384,128)
                         const float* __restrict__ b_ih, const float* __restrict__ b_hh,
                         float* __restrict__ hF,          // (R,128) in/out
                         __bf16* __restrict__ hBf,        // (R,128) in/out
                         int Rows) {
  int lane = threadIdx.x & 31;
  int wave = threadIdx.x >> 5;
  int rowTile = blockIdx.x * (blockDim.x >> 5) + wave;
  int row0 = rowTile * 16;
  if (row0 >= Rows) return;

  v16bf a1[8];  // aio operand, K=256
  v16bf a2[4];  // h operand,   K=128
#pragma unroll
  for (int ks = 0; ks < 8; ++ks) a1[ks] = load_a_frag(aio, 2 * kH, row0, ks, lane);
#pragma unroll
  for (int ks = 0; ks < 4; ++ks) a2[ks] = load_a_frag(hBf, kH, row0, ks, lane);

  int nl = lane & 15, hi = (lane >> 4) & 1;

  for (int ct = 0; ct < 8; ++ct) {   // column tile over H
    v8f gi[3], gh[3];
#pragma unroll
    for (int p = 0; p < 3; ++p) {    // r, i, n gate blocks
      v8f c = {};
#pragma unroll
      for (int ks = 0; ks < 8; ++ks) {
        v16bf bf = load_b_frag(wih, 2 * kH, p * 8 + ct, ks, lane, 3 * kH - 1);
        c = __builtin_amdgcn_wmma_f32_16x16x32_bf16(false, a1[ks], false, bf,
                                                    (short)0, c, false, false);
      }
      gi[p] = c;
      v8f d = {};
#pragma unroll
      for (int ks = 0; ks < 4; ++ks) {
        v16bf bf = load_b_frag(whh, kH, p * 8 + ct, ks, lane, 3 * kH - 1);
        d = __builtin_amdgcn_wmma_f32_16x16x32_bf16(false, a2[ks], false, bf,
                                                    (short)0, d, false, false);
      }
      gh[p] = d;
    }
    int col = ct * 16 + nl;
    float bi_r = b_ih[col], bi_i = b_ih[kH + col], bi_n = b_ih[2 * kH + col];
    float bh_r = b_hh[col], bh_i = b_hh[kH + col], bh_n = b_hh[2 * kH + col];
#pragma unroll
    for (int v = 0; v < 8; ++v) {
      int row = row0 + hi * 8 + v;
      float ir = gi[0][v] + bi_r, ii = gi[1][v] + bi_i, inn = gi[2][v] + bi_n;
      float hr = gh[0][v] + bh_r, hii = gh[1][v] + bh_i, hn = gh[2][v] + bh_n;
      float rg = sigmoidf_(ir + hr);
      float ig = sigmoidf_(ii + hii);
      float ng = tanhf(inn + rg * hn);
      float hv = hF[(size_t)row * kH + col];
      float nh = ng + ig * (hv - ng);
      hF[(size_t)row * kH + col]  = nh;       // each (row,col) touched by one lane
      hBf[(size_t)row * kH + col] = f2bf(nh); // a2 already loaded -> in-place safe
    }
  }
}

// ---- Attention + session vector (small: ~1.7 GFLOP, VALU + LDS) ----------
__global__ void attention(const float* __restrict__ hF,   // (B*50,128) new h
                          const int* __restrict__ alias_, // (B,50)
                          const int* __restrict__ mask,   // (B,50)
                          const float* __restrict__ fc1w, const float* __restrict__ fc1b,
                          const float* __restrict__ fc2w, const float* __restrict__ fc2b,
                          const float* __restrict__ fc3w,
                          float* __restrict__ aF, __bf16* __restrict__ aB) {
  __shared__ float sh[kS][kH];
  __shared__ float red[4];
  int b = blockIdx.x;
  int t = threadIdx.x;          // 0..127 = hidden column
  int lane = t & 31, wave = t >> 5;

  int msum = 0;
  for (int s = 0; s < kS; ++s) msum += mask[b * kS + s];
  int last = msum - 1; if (last < 0) last = 0;

  for (int s = 0; s < kS; ++s) {
    int al = alias_[b * kS + s];
    sh[s][t] = hF[((size_t)b * kN + al) * kH + t];
  }
  __syncthreads();

  float q1 = fc1b[t];
  for (int k = 0; k < kH; ++k) q1 += fc1w[t * kH + k] * sh[last][k];

  float acc = 0.f;
  float f3 = fc3w[t];
  float b2 = fc2b[t];
  for (int s = 0; s < kS; ++s) {
    float q2 = b2;
    for (int k = 0; k < kH; ++k) q2 += fc2w[t * kH + k] * sh[s][k];
    float val = f3 * sigmoidf_(q1 + q2);
    for (int off = 16; off > 0; off >>= 1) val += __shfl_xor(val, off, 32);
    if (lane == 0) red[wave] = val;
    __syncthreads();
    float alpha = red[0] + red[1] + red[2] + red[3];
    __syncthreads();
    acc += alpha * sh[s][t] * (float)mask[b * kS + s];
  }
  aF[(size_t)b * kH + t] = acc;
  aB[(size_t)b * kH + t] = f2bf(acc);
}

extern "C" void kernel_launch(void* const* d_in, const int* in_sizes, int n_in,
                              void* d_out, int out_size, void* d_ws, size_t ws_size,
                              hipStream_t stream) {
  (void)in_sizes; (void)n_in; (void)out_size; (void)ws_size;
  const int*   alias_inputs = (const int*)d_in[0];
  const float* A      = (const float*)d_in[1];
  const int*   items  = (const int*)d_in[2];
  const int*   mask   = (const int*)d_in[3];
  const float* emb    = (const float*)d_in[4];
  const float* ein_w  = (const float*)d_in[5];
  const float* ein_b  = (const float*)d_in[6];
  const float* eou_w  = (const float*)d_in[7];
  const float* eou_b  = (const float*)d_in[8];
  const float* w_ih   = (const float*)d_in[9];
  const float* w_hh   = (const float*)d_in[10];
  const float* b_ih   = (const float*)d_in[11];
  const float* b_hh   = (const float*)d_in[12];
  const float* b_iah  = (const float*)d_in[13];
  const float* b_oah  = (const float*)d_in[14];
  const float* fc1_w  = (const float*)d_in[15];
  const float* fc1_b  = (const float*)d_in[16];
  const float* fc2_w  = (const float*)d_in[17];
  const float* fc2_b  = (const float*)d_in[18];
  const float* fc3_w  = (const float*)d_in[19];

  char* wsp = (char*)d_ws;
  auto alloc = [&](size_t bytes) {
    char* p = wsp; wsp += (bytes + 255) & ~(size_t)255; return p;
  };
  const int rows = kB * kN;  // 51200
  __bf16* emb_bf  = (__bf16*)alloc((size_t)kV * kH * 2);         // 25.6 MB
  float*  h_f     = (float*) alloc((size_t)rows * kH * 4);       // 26.2 MB
  __bf16* h_bf    = (__bf16*)alloc((size_t)rows * kH * 2);       // 13.1 MB
  __bf16* aio_bf  = (__bf16*)alloc((size_t)rows * 2 * kH * 2);   // 26.2 MB
  __bf16* A_pad   = (__bf16*)alloc((size_t)kB * 2 * 64 * 64 * 2);   // 16.8 MB
  __bf16* hvT     = (__bf16*)alloc((size_t)kB * 2 * kH * 64 * 2);   // 33.5 MB
  __bf16* einw_bf = (__bf16*)alloc((size_t)kH * kH * 2);
  __bf16* eouw_bf = (__bf16*)alloc((size_t)kH * kH * 2);
  __bf16* wih_bf  = (__bf16*)alloc((size_t)3 * kH * 2 * kH * 2);
  __bf16* whh_bf  = (__bf16*)alloc((size_t)3 * kH * kH * 2);
  float*  a_f     = (float*) alloc((size_t)kB * kH * 4);
  __bf16* a_bf    = (__bf16*)alloc((size_t)kB * kH * 2);

  const int hvTStride = 2 * kH * 64;  // per-batch stride in hvT (elements)

  // 1) bf16 conversions + padded A
  cvt_bf16<<<2048, 256, 0, stream>>>(emb, emb_bf, (size_t)kV * kH);
  cvt_bf16<<<64,   256, 0, stream>>>(ein_w, einw_bf, (size_t)kH * kH);
  cvt_bf16<<<64,   256, 0, stream>>>(eou_w, eouw_bf, (size_t)kH * kH);
  cvt_bf16<<<96,   256, 0, stream>>>(w_ih, wih_bf, (size_t)3 * kH * 2 * kH);
  cvt_bf16<<<96,   256, 0, stream>>>(w_hh, whh_bf, (size_t)3 * kH * kH);
  pad_A<<<(kB * 2 * 64 * 64 + 255) / 256, 256, 0, stream>>>(A, A_pad);
  zero_u32<<<1024, 256, 0, stream>>>((unsigned*)hvT, (size_t)kB * 2 * kH * 64 / 2);

  // 2) h = emb[items]
  gather_h<<<(rows * (kH / 4) + 255) / 256, 256, 0, stream>>>(emb, items, h_f, h_bf, rows);

  // 3) hv = h@W^T+b, written straight into the graph-matmul B-operand layout
  dim3 g1(rows / 16, 1);
  wmma_gemm_xwt<4, 1, 1, false, true><<<g1, 256, 0, stream>>>(
      h_bf, einw_bf, ein_b, (float*)nullptr, hvT,            rows, kH, hvTStride);
  wmma_gemm_xwt<4, 1, 1, false, true><<<g1, 256, 0, stream>>>(
      h_bf, eouw_bf, eou_b, (float*)nullptr, hvT + kH * 64,  rows, kH, hvTStride);

  // 4) batched graph matmuls -> concatenated GRU input (bf16)
  wmma_graph_mm<<<kB, 256, 0, stream>>>(A_pad, hvT, b_iah, b_oah, aio_bf);

  // 5) fused GRU cell, h updated in place (f32 + bf16)
  wmma_gru<<<rows / 128, 256, 0, stream>>>(aio_bf, wih_bf, whh_bf, b_ih, b_hh,
                                           h_f, h_bf, rows);

  // 6) attention / session vector
  attention<<<kB, 128, 0, stream>>>(h_f, alias_inputs, mask, fc1_w, fc1_b,
                                    fc2_w, fc2_b, fc3_w, a_f, a_bf);

  // 7) scores = a @ emb[1:]^T  (M=1024, N=99999, K=128) — dominant GEMM.
  //    MT=4 row tiles register-resident (4x B reuse), NT=4 N-tiles per wave,
  //    non-temporal stores for the 410MB output, prefetch for W rows.
  const int Nout = kV - 1;
  const int ntiles = (Nout + 15) / 16;                 // 6250
  dim3 g2(kB / (16 * 4), (ntiles + 8 * 4 - 1) / (8 * 4));  // (16, 196)
  wmma_gemm_xwt<4, 4, 4, true, false><<<g2, 256, 0, stream>>>(
      a_bf, emb_bf + kH, (const float*)nullptr, (float*)d_out, (__bf16*)nullptr,
      kB, Nout, 0);
}
